// MultiHeadAttention_3728031613102
// MI455X (gfx1250) — compile-verified, matrix operand order checked by reading, not run on hardware
//
#include <hip/hip_runtime.h>
#include <hip/hip_bf16.h>

// ---------------------------------------------------------------------------
// Types for CDNA5 WMMA (wave32): A/B = 16 bf16 per lane, C/D = 8 f32 per lane
// ---------------------------------------------------------------------------
typedef __attribute__((ext_vector_type(16))) __bf16         v16bf;
typedef __attribute__((ext_vector_type(8)))  float          v8f;
typedef __attribute__((ext_vector_type(16))) unsigned short v16us;
typedef __attribute__((ext_vector_type(8)))  unsigned short v8us;
typedef __attribute__((ext_vector_type(4)))  unsigned short v4us;
typedef __attribute__((ext_vector_type(4)))  int            v4i;

__device__ __forceinline__ unsigned short f2bf(float f) {
  unsigned int u = __float_as_uint(f);
  unsigned int r = u + 0x7FFFu + ((u >> 16) & 1u);   // round-to-nearest-even
  return (unsigned short)(r >> 16);
}

__device__ __forceinline__ v8f wmma_bf16(v16bf a, v16bf b, v8f c) {
  // (neg_a, A, neg_b, B, c_mod, C, reuse_a, reuse_b)
  return __builtin_amdgcn_wmma_f32_16x16x32_bf16(false, a, false, b, (short)0, c,
                                                 false, false);
}

// Load one 16x32 bf16 fragment from a row-major LDS tile.
// ISA layout: lanes 0-15 -> K 0..7 & 16..23 ; lanes 16-31 -> K 8..15 & 24..31
// => two contiguous 16-byte chunks per lane (ds_load_b128 x2).
__device__ __forceinline__ v16bf ld_frag(const unsigned short* base, int stride) {
  const int lane = threadIdx.x & 31;
  const unsigned short* p = base + (lane & 15) * stride + ((lane >> 4) << 3);
  v8us lo = *reinterpret_cast<const v8us*>(p);
  v8us hi = *reinterpret_cast<const v8us*>(p + 16);
  union { v16us u; v16bf b; } f;
#pragma unroll
  for (int i = 0; i < 8; ++i) { f.u[i] = lo[i]; f.u[i + 8] = hi[i]; }
  return f.b;
}

// ---------------------------------------------------------------------------
// Async 16B global -> LDS copy (CDNA5 GLOBAL_LOAD_ASYNC_TO_LDS_B128, tracked
// by ASYNCcnt).  Builtin takes v4i pointers (per hipcc diagnostic): cast via
// integers (generic LDS address truncates to the 32-bit LDS offset).
// Falls back to a synchronous vector copy if the builtin is absent.
// ---------------------------------------------------------------------------
#if __has_builtin(__builtin_amdgcn_global_load_async_to_lds_b128)
#define ATHENA_HAVE_ASYNC_LDS 1
typedef __attribute__((address_space(1))) v4i as1_v4i;
typedef __attribute__((address_space(3))) v4i as3_v4i;
#endif

__device__ __forceinline__ void cp16_g2l(const unsigned short* g, unsigned short* l) {
#ifdef ATHENA_HAVE_ASYNC_LDS
  __builtin_amdgcn_global_load_async_to_lds_b128(
      (as1_v4i*)(size_t)g,
      (as3_v4i*)(unsigned int)(size_t)l, 0, 0);
#else
  *reinterpret_cast<v8us*>(l) = *reinterpret_cast<const v8us*>(g);
#endif
}

__device__ __forceinline__ void wait_async_lds() {
#ifdef ATHENA_HAVE_ASYNC_LDS
#if __has_builtin(__builtin_amdgcn_s_wait_asynccnt)
  __builtin_amdgcn_s_wait_asynccnt(0);
#else
  asm volatile("s_wait_asynccnt 0x0" ::: "memory");
#endif
#endif
}

// ---------------------------------------------------------------------------
// One-shot f32 -> bf16 conversion (memory-bound pre-pass; keeps the GEMM hot
// loop free of conversion VALU work).  n must be a multiple of 1024.
// ---------------------------------------------------------------------------
__global__ __launch_bounds__(256) void cvt_bf16(const float* __restrict__ in,
                                                unsigned short* __restrict__ out,
                                                int n4) {
  int i = blockIdx.x * 256 + threadIdx.x;
  if (i < n4) {
    float4 f = reinterpret_cast<const float4*>(in)[i];
    v4us p = { f2bf(f.x), f2bf(f.y), f2bf(f.z), f2bf(f.w) };
    reinterpret_cast<v4us*>(out)[i] = p;
  }
}

// ---------------------------------------------------------------------------
// GEMM: C[M,N] = A[M,K] * B[N,K]^T  (bf16 in, f32 accum/out)
// Block 256 (8 waves). Tile 128x128, BK=32, double-buffered LDS with async
// global->LDS copies.  Wave w: m-strip (w&3)*32, n-strip (w>>2)*64.
// Requires M%128==0, N%128==0, K%32==0.
// ---------------------------------------------------------------------------
#define LSTR 48  // padded bf16 row stride (96B; fragment chunks stay 16B aligned)

__global__ __launch_bounds__(256) void gemm_bf16t(const unsigned short* __restrict__ A,
                                                  const unsigned short* __restrict__ B,
                                                  float* __restrict__ C,
                                                  int N, int K) {
  __shared__ unsigned short As[2][128 * LSTR];
  __shared__ unsigned short Bs[2][128 * LSTR];

  const int tid  = threadIdx.x;
  const int lane = tid & 31;
  const int w    = tid >> 5;
  const int wm   = w & 3;
  const int wn   = w >> 2;
  const int m0   = blockIdx.y * 128;
  const int n0   = blockIdx.x * 128;

  const v8f vzero = {0.f, 0.f, 0.f, 0.f, 0.f, 0.f, 0.f, 0.f};
  v8f acc[2][4];
#pragma unroll
  for (int mi = 0; mi < 2; ++mi)
#pragma unroll
    for (int ni = 0; ni < 4; ++ni) acc[mi][ni] = vzero;

  // each thread copies 2 x 16B per matrix per tile (128 rows x 32 bf16)
  auto load_tile = [&](int buf, int k0) {
#pragma unroll
    for (int i = 0; i < 2; ++i) {
      int v = tid + i * 256;        // 512 chunks per matrix
      int r = v >> 2;               // row 0..127
      int c = (v & 3) << 3;         // bf16 col 0,8,16,24
      cp16_g2l(&A[(size_t)(m0 + r) * K + k0 + c], &As[buf][r * LSTR + c]);
      cp16_g2l(&B[(size_t)(n0 + r) * K + k0 + c], &Bs[buf][r * LSTR + c]);
    }
  };

  const int T = K >> 5;
  load_tile(0, 0);

  for (int t = 0; t < T; ++t) {
    const int p = t & 1;
    wait_async_lds();               // my wave's tile-t copies landed in LDS
    __syncthreads();                // everyone's landed; buf 1-p free to reuse
    if (t + 1 < T) load_tile(1 - p, (t + 1) << 5);
    if (t + 2 < T) {                // prefetch tile t+2 into caches
      int v = tid & 511;
      int r = v >> 2, c = (v & 3) << 3;
      __builtin_prefetch(&A[(size_t)(m0 + r) * K + ((t + 2) << 5) + c], 0, 1);
      __builtin_prefetch(&B[(size_t)(n0 + r) * K + ((t + 2) << 5) + c], 0, 1);
    }

    v16bf af[2], bf[4];
#pragma unroll
    for (int mi = 0; mi < 2; ++mi) af[mi] = ld_frag(&As[p][(wm * 32 + mi * 16) * LSTR], LSTR);
#pragma unroll
    for (int ni = 0; ni < 4; ++ni) bf[ni] = ld_frag(&Bs[p][(wn * 64 + ni * 16) * LSTR], LSTR);
#pragma unroll
    for (int mi = 0; mi < 2; ++mi)
#pragma unroll
      for (int ni = 0; ni < 4; ++ni) acc[mi][ni] = wmma_bf16(af[mi], bf[ni], acc[mi][ni]);
  }

  // C/D layout: VGPR i -> M = i (lanes 0-15) / i+8 (lanes 16-31); N = lane&15
  const int half8 = (lane >> 4) << 3;
  const int nlo   = lane & 15;
  float* cp = C + (size_t)(m0 + wm * 32 + half8) * N + (n0 + wn * 64 + nlo);
#pragma unroll
  for (int mi = 0; mi < 2; ++mi)
#pragma unroll
    for (int ni = 0; ni < 4; ++ni)
#pragma unroll
      for (int i = 0; i < 8; ++i)
        cp[(size_t)(mi * 16 + i) * N + ni * 16] = acc[mi][ni][i];
}

// ---------------------------------------------------------------------------
// RoPE + split into head-major bf16 q/k/v.  qkv: [B*S, 6144] fp32.
// q gets sqrt(128)*attn_scale[h] folded in.  Output: [B,H,S,128] bf16.
// ---------------------------------------------------------------------------
__global__ __launch_bounds__(256) void rope_split(const float* __restrict__ qkv,
                                                  const float* __restrict__ attn_scale,
                                                  unsigned short* __restrict__ qb,
                                                  unsigned short* __restrict__ kb,
                                                  unsigned short* __restrict__ vb) {
  const int gid = blockIdx.x * 256 + threadIdx.x;   // B*S*2048 total, exact
  const int c = gid & 2047;
  const int s = (gid >> 11) & 2047;
  const int b = gid >> 22;
  const int h = c >> 7;
  const int d = c & 127;

  const size_t base = (size_t)(b * 2048 + s) * 6144;
  const float qv = qkv[base + c];
  const float kv = qkv[base + 2048 + c];
  const float vv = qkv[base + 4096 + c];

  // inv_freq = 10000^(-(d&63)/64);  emb = concat([freqs,freqs])
  const int   f   = d & 63;
  const float inv = __expf(-(float)f * (9.210340371976184f / 64.f)); // ln(1e4)
  const float ang = (float)s * inv;
  const float cs  = __cosf(ang);
  const float sn  = __sinf(ang);

  // rotate_half: rh[d] = d<64 ? -x[2d+1] : x[2(d-64)]
  const int   dp  = (d < 64) ? (2 * d + 1) : (2 * (d - 64));
  const int   cp  = h * 128 + dp;
  const float qp  = qkv[base + cp];
  const float kp  = qkv[base + 2048 + cp];
  const float rhq = (d < 64) ? -qp : qp;
  const float rhk = (d < 64) ? -kp : kp;

  const float qo = (qv * cs + rhq * sn) * (11.313708498984761f * attn_scale[h]);
  const float ko = (kv * cs + rhk * sn);

  const size_t o = (((size_t)b * 16 + h) * 2048 + s) * 128 + d;
  qb[o] = f2bf(qo);
  kb[o] = f2bf(ko);
  vb[o] = f2bf(vv);
}

// ---------------------------------------------------------------------------
// Flash-style causal attention.  Grid: (S/64, H, B), block 256 (8 waves).
// Q block 64x128, kv tiles of 32.  Scores via WMMA -> LDS -> online softmax
// (one thread per row) -> P(bf16) @ V via WMMA, accumulators in registers.
// Output written as bf16 (feeds the bf16 output-projection GEMM directly).
// ---------------------------------------------------------------------------
#define QSTR 136   // bf16 row stride for 128-wide tiles (272B, 16B aligned)
#define PSTR 48    // bf16 row stride for 32-wide tiles (96B)
#define SSTR 36    // f32 score row stride

__global__ __launch_bounds__(256) void attn_kernel(const unsigned short* __restrict__ qb,
                                                   const unsigned short* __restrict__ kb,
                                                   const unsigned short* __restrict__ vb,
                                                   unsigned short* __restrict__ out) {
  const int S = 2048, H = 16, DH = 128;
  __shared__ unsigned short Qs[64 * QSTR];
  __shared__ unsigned short Ks[32 * QSTR];
  __shared__ unsigned short Vt[128 * PSTR];   // V transposed: [d][kv]
  __shared__ unsigned short Ps[64 * PSTR];    // exp probs, bf16
  __shared__ float Sc[64 * SSTR];
  __shared__ float m_s[64], l_s[64], a_s[64];

  const int tid  = threadIdx.x;
  const int lane = tid & 31;
  const int w    = tid >> 5;
  const int wm   = w & 3;     // q 16-row tile within 64
  const int wn2  = w >> 2;    // kv 16-col tile within 32 (scores) / n half (PV)
  const int half8 = (lane >> 4) << 3;
  const int nlo   = lane & 15;

  const int q0 = blockIdx.x * 64;
  const int h  = blockIdx.y;
  const int b  = blockIdx.z;
  const size_t hbase = ((size_t)b * H + h) * (size_t)S * DH;

  // stage Q block (64 rows x 128 bf16) -> LDS
#pragma unroll
  for (int i = 0; i < 4; ++i) {
    int v = tid + i * 256;                    // 1024 vec8 units
    int r = v >> 4;
    int c = (v & 15) << 3;
    *reinterpret_cast<v8us*>(&Qs[r * QSTR + c]) =
        *reinterpret_cast<const v8us*>(&qb[hbase + (size_t)(q0 + r) * DH + c]);
  }
  if (tid < 64) { m_s[tid] = -1e30f; l_s[tid] = 0.f; }

  const v8f vzero = {0.f, 0.f, 0.f, 0.f, 0.f, 0.f, 0.f, 0.f};
  v8f o[4];
#pragma unroll
  for (int j = 0; j < 4; ++j) o[j] = vzero;

  const int nkv = 2 * (blockIdx.x + 1);       // causal: kv0 <= q0+63
  for (int kt = 0; kt < nkv; ++kt) {
    const int kv0 = kt * 32;
    __syncthreads();                          // prior reads of Ks/Vt/Ps done
    // K tile 32x128
#pragma unroll
    for (int i = 0; i < 2; ++i) {
      int v = tid + i * 256;
      int r = v >> 4;
      int c = (v & 15) << 3;
      *reinterpret_cast<v8us*>(&Ks[r * QSTR + c]) =
          *reinterpret_cast<const v8us*>(&kb[hbase + (size_t)(kv0 + r) * DH + c]);
    }
    // V tile transposed into Vt[d][kv]
    for (int e = tid; e < 32 * 128; e += 256) {
      int r = e >> 7;
      int d = e & 127;
      Vt[d * PSTR + r] = vb[hbase + (size_t)(kv0 + r) * DH + d];
    }
    __syncthreads();

    // scores: 64x32 = 4 m-tiles x 2 n-tiles, one per wave, K=128 (4 steps)
    v8f s = vzero;
#pragma unroll
    for (int ks = 0; ks < 4; ++ks) {
      v16bf aq = ld_frag(&Qs[(wm * 16) * QSTR + ks * 32], QSTR);
      v16bf bk = ld_frag(&Ks[(wn2 * 16) * QSTR + ks * 32], QSTR);
      s = wmma_bf16(aq, bk, s);
    }
    // causal mask + dump scores to LDS
    const int coln = kv0 + wn2 * 16 + nlo;
#pragma unroll
    for (int i = 0; i < 8; ++i) {
      int rowl = wm * 16 + half8 + i;
      float sv = s[i];
      if (coln > q0 + rowl) sv = -1e9f;
      Sc[rowl * SSTR + wn2 * 16 + nlo] = sv;
    }
    __syncthreads();

    // online softmax: one thread per q row
    if (tid < 64) {
      float mo = m_s[tid];
      float mx = mo;
      float r[32];
#pragma unroll
      for (int j = 0; j < 32; ++j) { r[j] = Sc[tid * SSTR + j]; mx = fmaxf(mx, r[j]); }
      float al = __expf(mo - mx);
      float sum = 0.f;
#pragma unroll
      for (int j = 0; j < 32; ++j) {
        float p = __expf(r[j] - mx);
        sum += p;
        Ps[tid * PSTR + j] = f2bf(p);
      }
      l_s[tid] = al * l_s[tid] + sum;
      m_s[tid] = mx;
      a_s[tid] = al;
    }
    __syncthreads();

    // rescale running output, then O += P(64x32) @ V(32x128)
    float al8[8];
#pragma unroll
    for (int i = 0; i < 8; ++i) al8[i] = a_s[wm * 16 + half8 + i];
#pragma unroll
    for (int j = 0; j < 4; ++j)
#pragma unroll
      for (int i = 0; i < 8; ++i) o[j][i] *= al8[i];

    v16bf ap = ld_frag(&Ps[(wm * 16) * PSTR], PSTR);
#pragma unroll
    for (int j = 0; j < 4; ++j) {
      v16bf bv = ld_frag(&Vt[((wn2 * 4 + j) * 16) * PSTR], PSTR);
      o[j] = wmma_bf16(ap, bv, o[j]);
    }
  }
  __syncthreads();

  float li[8];
#pragma unroll
  for (int i = 0; i < 8; ++i) li[i] = 1.f / l_s[wm * 16 + half8 + i];
#pragma unroll
  for (int j = 0; j < 4; ++j) {
    int dcol = (wn2 * 4 + j) * 16 + nlo;
#pragma unroll
    for (int i = 0; i < 8; ++i) {
      int rowl = wm * 16 + half8 + i;
      out[(size_t)(b * S + q0 + rowl) * 2048 + h * 128 + dcol] = f2bf(o[j][i] * li[i]);
    }
  }
}

// ---------------------------------------------------------------------------
// Launch: cvt(x,w_qkv,w_out) -> qkv GEMM -> rope/split -> attention -> out proj
// inputs: 0:x  1:mask(unused; causal tril applied analytically)
//         2:w_qkv[6144,2048]  3:w_out[2048,2048]  4:attn_scale[16]
// ---------------------------------------------------------------------------
extern "C" void kernel_launch(void* const* d_in, const int* in_sizes, int n_in,
                              void* d_out, int out_size, void* d_ws, size_t ws_size,
                              hipStream_t stream) {
  (void)in_sizes; (void)n_in; (void)out_size; (void)ws_size;
  const float* x          = (const float*)d_in[0];
  const float* w_qkv      = (const float*)d_in[2];
  const float* w_out      = (const float*)d_in[3];
  const float* attn_scale = (const float*)d_in[4];

  char* ws = (char*)d_ws;
  float*          qkv    = (float*)ws;                          // 4096*6144*4  = 100663296 B
  unsigned short* qb     = (unsigned short*)(ws + 100663296);   // 8388608 elems
  unsigned short* kb     = qb + 8388608;
  unsigned short* vb     = kb + 8388608;
  unsigned short* attnb  = (unsigned short*)(ws + 150994944);   // 4096*2048 bf16
  unsigned short* xb     = (unsigned short*)(ws + 167772160);   // 4096*2048 bf16
  unsigned short* wqkvb  = (unsigned short*)(ws + 184549376);   // 6144*2048 bf16
  unsigned short* woutb  = (unsigned short*)(ws + 209715200);   // 2048*2048 bf16

  // 0) one-shot f32 -> bf16 operand conversion (memory-bound)
  cvt_bf16<<<8192,  256, 0, stream>>>(x,     xb,    2097152);   // 8388608  elems
  cvt_bf16<<<12288, 256, 0, stream>>>(w_qkv, wqkvb, 3145728);   // 12582912 elems
  cvt_bf16<<<4096,  256, 0, stream>>>(w_out, woutb, 1048576);   // 4194304  elems

  // 1) qkv = x @ w_qkv^T        (M=4096, N=6144, K=2048)
  gemm_bf16t<<<dim3(48, 32), 256, 0, stream>>>(xb, wqkvb, qkv, 6144, 2048);
  // 2) RoPE + scale + split to head-major bf16
  rope_split<<<32768, 256, 0, stream>>>(qkv, attn_scale, qb, kb, vb);
  // 3) causal flash attention   grid (S/64, H, B)  -> bf16
  attn_kernel<<<dim3(32, 16, 2), 256, 0, stream>>>(qb, kb, vb, attnb);
  // 4) out = attn @ w_out^T     (M=4096, N=2048, K=2048)
  gemm_bf16t<<<dim3(16, 32), 256, 0, stream>>>(attnb, woutb, (float*)d_out, 2048, 2048);
}
